// NetVLAD_58188216926562
// MI455X (gfx1250) — matrix operand order repaired and checked by reading, not compiled, not run on hardware
//
#include <hip/hip_runtime.h>
#include <hip/hip_bf16.h>

#define B_ 32
#define D_ 512
#define T_ 64
#define R_ 49
#define K_ 64
#define TR_ 3136
#define OUT_ 1024
#define KD_ 32768
#define TRTILE 32
#define NTILES 98   // 3136 / 32

typedef __bf16 bf16_t;
typedef __attribute__((ext_vector_type(16))) __bf16 v16bf;
typedef __attribute__((ext_vector_type(8)))  float  v8f;
typedef __attribute__((ext_vector_type(4)))  unsigned v4u;
typedef __attribute__((ext_vector_type(8)))  int      v8i;
typedef __attribute__((ext_vector_type(4)))  int      v4i;

// Hardware f32 -> bf16 conversion (v_cvt, 1 VALU op, pairable by the compiler).
__device__ __forceinline__ bf16_t f2bf(float f) { return (bf16_t)f; }

__device__ __forceinline__ v8f vzero8() {
  v8f z;
#pragma unroll
  for (int i = 0; i < 8; ++i) z[i] = 0.f;
  return z;
}

__device__ __forceinline__ v8f wmma_bf16(v16bf a, v16bf b, v8f c) {
  return __builtin_amdgcn_wmma_f32_16x16x32_bf16(false, a, false, b, (short)0, c,
                                                 false, false);
}

// A fragment 16(M)x32(K) from bf16 row-major source, leading dim ld.
__device__ __forceinline__ v16bf frag_a_bf16(const bf16_t* __restrict__ src, int ld,
                                             int lane) {
  v16bf a;
  const int r = lane & 15, h = lane >> 4;
  const bf16_t* p = src + r * ld;
#pragma unroll
  for (int i = 0; i < 4; ++i) {
    const int k = 16 * h + 2 * i;
    a[2 * i] = p[k];
    a[2 * i + 1] = p[k + 1];
  }
#pragma unroll
  for (int i = 4; i < 8; ++i) {
    const int k = 16 * h + 2 * i;
    a[2 * i] = p[k];
    a[2 * i + 1] = p[k + 1];
  }
  return a;
}

// A fragment 16x32 from f32 row-major source (converted to bf16 inline).
__device__ __forceinline__ v16bf frag_a_f32(const float* __restrict__ src, int ld,
                                            int lane) {
  v16bf a;
  const int r = lane & 15, h = lane >> 4;
  const float* p = src + r * ld;
#pragma unroll
  for (int i = 0; i < 4; ++i) {
    const int k = 16 * h + 2 * i;
    a[2 * i] = f2bf(p[k]);
    a[2 * i + 1] = f2bf(p[k + 1]);
  }
#pragma unroll
  for (int i = 4; i < 8; ++i) {
    const int k = 16 * h + 2 * i;
    a[2 * i] = f2bf(p[k]);
    a[2 * i + 1] = f2bf(p[k + 1]);
  }
  return a;
}

// B fragment 32(K)x16(N) from bf16 row-major source, leading dim ld.
__device__ __forceinline__ v16bf frag_b_bf16(const bf16_t* __restrict__ src, int ld,
                                             int lane) {
  v16bf b;
  const int n = lane & 15, h = lane >> 4;
#pragma unroll
  for (int j = 0; j < 8; ++j) {
    const int k = 16 * h + 2 * j;
    b[2 * j] = src[k * ld + n];
    b[2 * j + 1] = src[(k + 1) * ld + n];
  }
  return b;
}

// B fragment 32x16 where B[kk][n] = src[n*ld + kk] (transposed bf16 source):
// each lane reads 16 consecutive bf16 (32 B) -> two ds_load_b128.
__device__ __forceinline__ v16bf frag_bT_bf16(const bf16_t* __restrict__ src, int ld,
                                              int lane) {
  v16bf b;
  const int n = lane & 15, h = lane >> 4;
  const bf16_t* p = src + (size_t)n * ld + 16 * h;
#pragma unroll
  for (int j = 0; j < 16; ++j) b[j] = p[j];
  return b;
}

// ---- Tensor Data Mover: async-load a 2D fp32 tile (512 rows x 32 cols,
// row stride 3136 elems) from global into LDS. 6-arg clang-23 builtin. ----
__device__ __forceinline__ void tdm_load_tile(unsigned lds_off,
                                              const float* __restrict__ gptr) {
  const unsigned long long ga = (unsigned long long)(const void*)gptr;
  v4u g0;
  g0[0] = 1u;                                              // count=1, user D#
  g0[1] = lds_off;                                         // lds_addr
  g0[2] = (unsigned)ga;                                    // global_addr lo
  g0[3] = (unsigned)((ga >> 32) & 0x01FFFFFFu) | (2u << 30);  // addr hi | type=2
  v8i g1;
  g1[0] = (int)(2u << 16);              // wg_mask=0, data_size=4B
  g1[1] = (int)((unsigned)TR_ << 16);   // tensor_dim0 lo16 (3136)
  g1[2] = (int)((unsigned)D_ << 16);    // dim0 hi16=0 | tensor_dim1 lo16 (512)
  g1[3] = (int)((unsigned)TRTILE << 16);// dim1 hi16=0 | tile_dim0 (32)
  g1[4] = D_;                           // tile_dim1=512 | tile_dim2=0
  g1[5] = TR_;                          // tensor_dim0_stride lo32
  g1[6] = 0;
  g1[7] = 0;
  const v4i z4 = {0, 0, 0, 0};
  const v8i z8 = {0, 0, 0, 0, 0, 0, 0, 0};
  __builtin_amdgcn_tensor_load_to_lds(g0, g1, z4, z4, z8, 0);
}

// -------------------------------------------------------------------------
// Stage 1a: per (b, chunk): fused logits -> softmax -> partial VLAD.
// grid (32, nchunks), 256 threads (8 waves). TDM stages fp32 tile; one
// cooperative pass converts it to a bf16 copy that all fragments read.
// LDS: bufF f32[512][32] @0 (64K) | sXb bf16[512][32] @64K (32K) |
//      sL f32 64x34 | sA bf16 64x32 | sRed | pad | sAsum @128K.
// vbuf (64x512 f32, 128K) overlays [0,128K) in the epilogue only.
// -------------------------------------------------------------------------
#define OFF_XF 0u
#define OFF_XB 65536
#define OFF_SL 98304
#define OFF_SA 107008
#define OFF_SRED 111104
#define OFF_SASUM 131072
#define SMEM1_BYTES 131328

__global__ __launch_bounds__(256) void netvlad_stage1a(
    const float* __restrict__ x, const int* __restrict__ mask,
    const float* __restrict__ conv_w, float* __restrict__ part_vlad,
    float* __restrict__ part_asum, int nchunks, int tpc) {
  __shared__ __align__(16) unsigned char smem[SMEM1_BYTES];
  float*  bufF  = (float*)(smem + OFF_XF);     // fp32 TDM staging, 512x32
  bf16_t* sXb   = (bf16_t*)(smem + OFF_XB);    // bf16 tile copy, 512x32
  float*  sL    = (float*)(smem + OFF_SL);     // 64 x 34
  bf16_t* sA    = (bf16_t*)(smem + OFF_SA);    // 64 x 32
  float*  sRed  = (float*)(smem + OFF_SRED);   // 8 x 32
  float*  sAsum = (float*)(smem + OFF_SASUM);  // 64 (above epilogue overlay)

  const int tid  = threadIdx.x;
  const int lane = tid & 31;
  const int wid  = tid >> 5;
  const int b     = blockIdx.x;
  const int chunk = blockIdx.y;

  if (tid < K_) sAsum[tid] = 0.f;

  // Preload this wave's conv-weight A-fragments (tile-invariant): kt = wid>>1.
  const int kt_g1 = wid >> 1;
  v16bf wfrag[16];
  {
    const float* wp = conv_w + (size_t)(kt_g1 * 16) * D_;
#pragma unroll
    for (int ds = 0; ds < 16; ++ds) wfrag[ds] = frag_a_f32(wp + ds * 32, D_, lane);
  }

  // Persistent partial-VLAD accumulators: wave owns D cols [wid*64, wid*64+64).
  v8f vacc[4][4];
#pragma unroll
  for (int kt = 0; kt < 4; ++kt)
#pragma unroll
    for (int nt = 0; nt < 4; ++nt) vacc[kt][nt] = vzero8();

  const float* xb = x + (size_t)b * D_ * TR_;
  const int it0 = chunk * tpc;
  int it1 = it0 + tpc;
  if (it1 > NTILES) it1 = NTILES;

  if (wid == 0 && it0 < it1) tdm_load_tile(OFF_XF, xb + it0 * TRTILE);

  for (int it = it0; it < it1; ++it) {
    const int trb = it * TRTILE;

    if (wid == 0) __builtin_amdgcn_s_wait_tensorcnt((short)0);
    __syncthreads();  // fp32 tile landed for everyone

    // ---- convert fp32 tile -> bf16 copy (each element exactly once) ----
    for (int i = tid; i < (D_ * TRTILE) / 4; i += 256) {
      const float4 v = ((const float4*)bufF)[i];
      bf16_t* o = sXb + i * 4;
      o[0] = f2bf(v.x);
      o[1] = f2bf(v.y);
      o[2] = f2bf(v.z);
      o[3] = f2bf(v.w);
    }
    __syncthreads();  // sXb ready; bufF free for the next TDM

    if (wid == 0 && it + 1 < it1)
      tdm_load_tile(OFF_XF, xb + (it + 1) * TRTILE);  // overlaps compute below

    // ---- GEMM1: logits(64x32) = W(64x512) * X(512x32); 1 C tile per wave ----
    {
      const int trt = wid & 1;
      v8f c0 = vzero8();
#pragma unroll 4
      for (int ds = 0; ds < 16; ++ds) {
        const v16bf bf = frag_b_bf16(sXb + (ds * 32) * TRTILE + trt * 16, TRTILE, lane);
        c0 = wmma_bf16(wfrag[ds], bf, c0);
      }
      const int m0 = (lane >> 4) * 8;
      const int n  = lane & 15;
#pragma unroll
      for (int i = 0; i < 8; ++i)
        sL[(kt_g1 * 16 + m0 + i) * 34 + trt * 16 + n] = c0[i];
    }
    __syncthreads();

    // ---- masked softmax over k (64) per tr column (32 cols, 8 segments) ----
    {
      const int tr  = tid & 31;
      const int seg = tid >> 5;
      const int k0s = seg * 8;
      float lmax = -3.4e38f;
#pragma unroll
      for (int i = 0; i < 8; ++i) lmax = fmaxf(lmax, sL[(k0s + i) * 34 + tr]);
      sRed[seg * 32 + tr] = lmax;
      __syncthreads();
      float cmax = sRed[tr];
#pragma unroll
      for (int s = 1; s < 8; ++s) cmax = fmaxf(cmax, sRed[s * 32 + tr]);
      __syncthreads();
      float lsum = 0.f;
#pragma unroll
      for (int i = 0; i < 8; ++i) {
        const float e = __expf(sL[(k0s + i) * 34 + tr] - cmax);
        sL[(k0s + i) * 34 + tr] = e;
        lsum += e;
      }
      sRed[seg * 32 + tr] = lsum;
      __syncthreads();
      float tot = sRed[tr];
#pragma unroll
      for (int s = 1; s < 8; ++s) tot += sRed[s * 32 + tr];
      const float mval = (float)mask[b * T_ + (trb + tr) / R_];
      const float inv  = mval / tot;
#pragma unroll
      for (int i = 0; i < 8; ++i) {
        const float a = sL[(k0s + i) * 34 + tr] * inv;
        sL[(k0s + i) * 34 + tr] = a;
        sA[(k0s + i) * 32 + tr] = f2bf(a);
      }
    }
    __syncthreads();

    // ---- a_sum accumulate (one thread per k; alongside GEMM2) ----
    if (tid < K_) {
      float s = 0.f;
#pragma unroll 8
      for (int tr = 0; tr < 32; ++tr) s += sL[tid * 34 + tr];
      sAsum[tid] += s;
    }

    // ---- GEMM2: vlad(64x512) += A(64x32) * X^T(32x512), this wave's D slice ----
    {
      const int dbase = wid * 64;
      v16bf afr[4];
#pragma unroll
      for (int kt = 0; kt < 4; ++kt)
        afr[kt] = frag_a_bf16(sA + (kt * 16) * TRTILE, TRTILE, lane);
#pragma unroll
      for (int nt = 0; nt < 4; ++nt) {
        const v16bf bfr =
            frag_bT_bf16(sXb + (size_t)(dbase + nt * 16) * TRTILE, TRTILE, lane);
#pragma unroll
        for (int kt = 0; kt < 4; ++kt)
          vacc[kt][nt] = wmma_bf16(afr[kt], bfr, vacc[kt][nt]);
      }
    }
    __syncthreads();
  }

  // ---- epilogue: dump accumulators to LDS (overlay [0,128K)), write partials ----
  float* vbuf = (float*)smem;  // 64 x 512 f32 = 128K
  {
    const int m0 = (lane >> 4) * 8;
    const int n  = lane & 15;
    const int dbase = wid * 64;
#pragma unroll
    for (int kt = 0; kt < 4; ++kt)
#pragma unroll
      for (int nt = 0; nt < 4; ++nt)
#pragma unroll
        for (int i = 0; i < 8; ++i)
          vbuf[(kt * 16 + m0 + i) * D_ + dbase + nt * 16 + n] = vacc[kt][nt][i];
  }
  __syncthreads();
  float* pv = part_vlad + ((size_t)b * nchunks + chunk) * (K_ * D_);
  for (int i = tid; i < (K_ * D_) / 4; i += 256)
    ((float4*)pv)[i] = ((const float4*)vbuf)[i];
  if (tid < K_) part_asum[((size_t)b * nchunks + chunk) * K_ + tid] = sAsum[tid];
}

// -------------------------------------------------------------------------
// Stage 1b: reduce chunk partials, subtract a_sum*centroids, intra + global
// L2-normalize, emit bf16 vladn. grid 32, 256 threads.
// -------------------------------------------------------------------------
__global__ __launch_bounds__(256) void netvlad_stage1b(
    const float* __restrict__ part_vlad, const float* __restrict__ part_asum,
    const float* __restrict__ centroids, bf16_t* __restrict__ vladn, int nchunks) {
  __shared__ float sRed[256];
  __shared__ float sAs[64];
  __shared__ float sScal[64];
  __shared__ float sG[1];
  const int b = blockIdx.x, tid = threadIdx.x;
  const int k = tid >> 2, q = tid & 3;

  if (tid < K_) {
    float s = 0.f;
    for (int c = 0; c < nchunks; ++c)
      s += part_asum[((size_t)b * nchunks + c) * K_ + tid];
    sAs[tid] = s;
  }
  __syncthreads();

  const float as = sAs[k];
  const float* cr = centroids + (size_t)k * D_;
  float v[128];
  float ss = 0.f;
  for (int j = 0; j < 128; ++j) {
    const int d = q + 4 * j;
    float acc = 0.f;
    for (int c = 0; c < nchunks; ++c)
      acc += part_vlad[((size_t)b * nchunks + c) * (size_t)KD_ + k * D_ + d];
    acc -= as * cr[d];
    v[j] = acc;
    ss += acc * acc;
  }
  sRed[q * 64 + k] = ss;
  __syncthreads();
  if (tid < K_) {
    const float tot = sRed[tid] + sRed[64 + tid] + sRed[128 + tid] + sRed[192 + tid];
    const float inv = 1.f / fmaxf(sqrtf(tot), 1e-12f);
    sScal[tid] = inv;
    sAs[tid] = tot * inv * inv;  // row contribution to global sumsq
  }
  __syncthreads();
  if (tid == 0) {
    float g = 0.f;
    for (int kk = 0; kk < K_; ++kk) g += sAs[kk];
    sG[0] = 1.f / fmaxf(sqrtf(g), 1e-12f);
  }
  __syncthreads();
  const float sc = sScal[k] * sG[0];
  bf16_t* outp = vladn + (size_t)b * KD_ + k * D_;
  for (int j = 0; j < 128; ++j) outp[q + 4 * j] = f2bf(v[j] * sc);
}

// -------------------------------------------------------------------------
// Stage 2: out_part(4,32,1024) partial GEMM: vladn(32x32768 bf16) x red_w^T.
// 256 blocks = 64 n-blocks x 4 K-slices; 4 waves split the slice further.
// -------------------------------------------------------------------------
__global__ __launch_bounds__(128) void netvlad_stage2(
    const bf16_t* __restrict__ vladn, const float* __restrict__ red_w,
    float* __restrict__ out_part) {
  __shared__ float red[4 * 2 * 256];
  const int tid  = threadIdx.x;
  const int lane = tid & 31;
  const int w    = tid >> 5;
  const int nb   = blockIdx.x & 63;
  const int ks   = blockIdx.x >> 6;
  const int n0   = nb * 16;

  v8f acc0 = vzero8();
  v8f acc1 = vzero8();
  const int kbase = ks * (KD_ / 4) + w * (KD_ / 16);
  for (int itk = 0; itk < (KD_ / 16) / 32; ++itk) {  // 64 iterations
    const int k0 = kbase + itk * 32;
    v16bf bfr;
    {
      const int n = lane & 15, h = lane >> 4;
      const float* p = red_w + (size_t)(n0 + n) * KD_ + k0 + 16 * h;
#pragma unroll
      for (int j = 0; j < 16; ++j) bfr[j] = f2bf(p[j]);
    }
    const v16bf a0 = frag_a_bf16(vladn + k0, KD_, lane);
    const v16bf a1 = frag_a_bf16(vladn + (size_t)16 * KD_ + k0, KD_, lane);
    acc0 = wmma_bf16(a0, bfr, acc0);
    acc1 = wmma_bf16(a1, bfr, acc1);
  }
  {
    const int m0 = (lane >> 4) * 8;
    const int n  = lane & 15;
#pragma unroll
    for (int i = 0; i < 8; ++i) {
      red[(w * 2 + 0) * 256 + (m0 + i) * 16 + n] = acc0[i];
      red[(w * 2 + 1) * 256 + (m0 + i) * 16 + n] = acc1[i];
    }
  }
  __syncthreads();
  for (int idx = tid; idx < 512; idx += 128) {
    const int mt = idx >> 8;
    const int e  = idx & 255;
    const float s = red[(0 * 2 + mt) * 256 + e] + red[(1 * 2 + mt) * 256 + e] +
                    red[(2 * 2 + mt) * 256 + e] + red[(3 * 2 + mt) * 256 + e];
    const int brow = mt * 16 + (e >> 4);
    const int col  = n0 + (e & 15);
    out_part[(size_t)ks * B_ * OUT_ + (size_t)brow * OUT_ + col] = s;
  }
}

// -------------------------------------------------------------------------
// Stage 3: sum the 4 K-slice partials + LayerNorm -> d_out. grid 32.
// -------------------------------------------------------------------------
__global__ __launch_bounds__(256) void netvlad_stage3(
    const float* __restrict__ out_part, const float* __restrict__ gamma,
    const float* __restrict__ beta, float* __restrict__ out) {
  __shared__ float sred[256];
  const int b = blockIdx.x;
  const int tid = threadIdx.x;
  float v[4];
#pragma unroll
  for (int i = 0; i < 4; ++i) {
    const int c = tid + i * 256;
    v[i] = out_part[(size_t)0 * B_ * OUT_ + b * OUT_ + c] +
           out_part[(size_t)1 * B_ * OUT_ + b * OUT_ + c] +
           out_part[(size_t)2 * B_ * OUT_ + b * OUT_ + c] +
           out_part[(size_t)3 * B_ * OUT_ + b * OUT_ + c];
  }
  sred[tid] = v[0] + v[1] + v[2] + v[3];
  __syncthreads();
  for (int off = 128; off > 0; off >>= 1) {
    if (tid < off) sred[tid] += sred[tid + off];
    __syncthreads();
  }
  const float mu = sred[0] * (1.f / 1024.f);
  __syncthreads();
  float vs = 0.f;
#pragma unroll
  for (int i = 0; i < 4; ++i) {
    const float d = v[i] - mu;
    vs += d * d;
  }
  sred[tid] = vs;
  __syncthreads();
  for (int off = 128; off > 0; off >>= 1) {
    if (tid < off) sred[tid] += sred[tid + off];
    __syncthreads();
  }
  const float var  = sred[0] * (1.f / 1024.f);
  const float rinv = rsqrtf(var + 1e-5f);
#pragma unroll
  for (int i = 0; i < 4; ++i) {
    const int c = tid + i * 256;
    out[(size_t)b * OUT_ + c] = (v[i] - mu) * rinv * gamma[c] + beta[c];
  }
}

extern "C" void kernel_launch(void* const* d_in, const int* in_sizes, int n_in,
                              void* d_out, int out_size, void* d_ws, size_t ws_size,
                              hipStream_t stream) {
  const float* x         = (const float*)d_in[0];
  const int*   mask      = (const int*)d_in[1];
  const float* centroids = (const float*)d_in[2];
  const float* conv_w    = (const float*)d_in[3];
  const float* red_w     = (const float*)d_in[4];
  const float* gamma     = (const float*)d_in[5];
  const float* beta      = (const float*)d_in[6];
  float* out = (float*)d_out;

  char* ws = (char*)d_ws;
  bf16_t* vladn   = (bf16_t*)ws;                               // 2 MB
  float* out_part = (float*)(ws + (size_t)(2u << 20));         // 512 KB
  size_t off = (size_t)(2u << 20) + (size_t)4 * B_ * OUT_ * sizeof(float);

  int nchunks = 7;  // 224 workgroups in stage 1a; fallback if ws is small
  {
    const size_t need = off + (size_t)nchunks * B_ * K_ * 4 +
                        (size_t)nchunks * B_ * (size_t)KD_ * 4;
    if (ws_size < need) nchunks = 1;
  }
  float* part_asum = (float*)(ws + off);
  float* part_vlad = (float*)(ws + off + (size_t)nchunks * B_ * K_ * 4);
  const int tpc = (NTILES + nchunks - 1) / nchunks;

  dim3 g1(B_, nchunks);
  netvlad_stage1a<<<g1, 256, 0, stream>>>(x, mask, conv_w, part_vlad, part_asum,
                                          nchunks, tpc);
  netvlad_stage1b<<<B_, 256, 0, stream>>>(part_vlad, part_asum, centroids, vladn,
                                          nchunks);
  netvlad_stage2<<<4 * (OUT_ / 16), 128, 0, stream>>>(vladn, red_w, out_part);
  netvlad_stage3<<<B_, 256, 0, stream>>>(out_part, gamma, beta, out);
}
